// KernelRegressionLayer_22720376995867
// MI455X (gfx1250) — matrix operand order repaired and checked by reading, not compiled, not run on hardware
//
#include <hip/hip_runtime.h>
#include <hip/hip_bf16.h>

// ---------------------------------------------------------------------------
// KernelRegressionLayer on MI455X (gfx1250, wave32).
//
//   out[i] = sum_{j<i} exp(-0.5*(xsq[i] + Xsq[j] - 2*x_i.X_j)) * y[j]
//
// Working set (16 MB fp32 + 16 MB bf16 split) is L2-resident (192 MB L2,
// 23.3 TB/s HBM => memory is a non-issue); the cost is the masked
// 8192x8192x256 GEMM. The bf16 matrix pipe has 8x the K-depth per
// instruction of the fp32 pipe (16x16x32 vs 16x16x4), so we use the
// split-precision 3xbf16 scheme (ah*bh + ah*bl + al*bh) -> ~2.7x fewer
// matrix-pipe cycles than exact fp32 WMMA at ~1e-4 absolute error on d2.
// Precomputed hi/lo split lives in d_ws; exact-fp32 WMMA fallback if d_ws
// is too small. Triangular mask culls ~half the blocks at launch.
// ---------------------------------------------------------------------------

typedef __attribute__((ext_vector_type(2)))  float  v2f;
typedef __attribute__((ext_vector_type(8)))  float  v8f;
typedef __attribute__((ext_vector_type(16))) __bf16 v16bf;

#define M_DIM 8192
#define N_DIM 8192
#define D_DIM 256

union Frag16 {           // one 16-bit WMMA operand: 16 x bf16 = 8 VGPRs
  uint4 u[2];
  v16bf v;
};

__global__ void zero_f32(float* p, int n) {
  int i = blockIdx.x * blockDim.x + threadIdx.x;
  if (i < n) p[i] = 0.0f;
}

// sq[r] = sum_d a[r][d]^2 in exact fp32 (tiny, L2-resident).
__global__ void row_sqnorm(const float* __restrict__ a, float* __restrict__ sq,
                           int rows) {
  int r = blockIdx.x * blockDim.x + threadIdx.x;
  if (r >= rows) return;
  const float4* p = reinterpret_cast<const float4*>(a + (size_t)r * D_DIM);
  float s = 0.0f;
#pragma unroll 8
  for (int k = 0; k < D_DIM / 4; ++k) {
    float4 v = p[k];
    s += v.x * v.x + v.y * v.y + v.z * v.z + v.w * v.w;
  }
  sq[r] = s;
}

// Split fp32 -> (hi, lo) bf16 pair: v ~= hi + lo to ~16 mantissa bits.
__global__ void split_bf16(const float* __restrict__ src,
                           __hip_bfloat16* __restrict__ hi,
                           __hip_bfloat16* __restrict__ lo, int n) {
  int i = blockIdx.x * blockDim.x + threadIdx.x;
  if (i >= n) return;
  float v = src[i];
  __hip_bfloat16 h = __float2bfloat16(v);
  hi[i] = h;
  lo[i] = __float2bfloat16(v - __bfloat162float(h));
}

// ===========================================================================
// Primary path: split-precision 3xbf16 WMMA (v_wmma_f32_16x16x32_bf16).
// Block = 256 threads = 8 wave32 waves.
//   blockIdx.x -> row strip of 16 rows
//   blockIdx.y -> column chunk of 512 (each wave owns 4 column tiles of 16)
// ===========================================================================
__global__ __launch_bounds__(256) void krr_wmma_bf16x3(
    const __hip_bfloat16* __restrict__ xh, const __hip_bfloat16* __restrict__ xl,
    const __hip_bfloat16* __restrict__ Xh, const __hip_bfloat16* __restrict__ Xl,
    const float* __restrict__ y, const float* __restrict__ xsq,
    const float* __restrict__ Xsq, float* __restrict__ out) {
  const int rowBase   = blockIdx.x * 16;
  const int chunkBase = blockIdx.y * 512;
  if (chunkBase >= rowBase + 16) return;  // fully above/at diagonal -> masked

  __shared__ float srow[16];
  const int tid = threadIdx.x;
  if (tid < 16) srow[tid] = 0.0f;
  __syncthreads();

  const int wave = tid >> 5;
  const int lane = tid & 31;
  const int half = lane >> 4;   // 0: lanes 0-15, 1: lanes 16-31
  const int lr   = lane & 15;
  const int ct0  = chunkBase + wave * 64;  // this wave's 4-tile column base

  if (ct0 < rowBase + 16) {  // wave-uniform: EXEC all-ones around WMMA
    // A 16x32 bf16 fragment (ISA 7.12.2): lane -> row M = lane%16;
    //   half 0 regs hold K = [k, k+8) and [k+16, k+24)
    //   half 1 regs hold K = [k+8, k+16) and [k+24, k+32)
    // B 32x16 bf16 fragment: lane -> col N = lane%16;
    //   half h regs hold K = [k + 8h, k + 8h + 8) and [k + 16 + 8h, k + 24 + 8h), same as A.
    const __hip_bfloat16* aH = xh + (size_t)(rowBase + lr) * D_DIM;
    const __hip_bfloat16* aL = xl + (size_t)(rowBase + lr) * D_DIM;

    v8f acc[4];
#pragma unroll
    for (int t = 0; t < 4; ++t) acc[t] = (v8f){};

#pragma unroll 2
    for (int k = 0; k < D_DIM; k += 32) {
      Frag16 Ah, Al;
      const int a0 = k + half * 8;
      const int a1 = k + 16 + half * 8;
      Ah.u[0] = *reinterpret_cast<const uint4*>(aH + a0);
      Ah.u[1] = *reinterpret_cast<const uint4*>(aH + a1);
      Al.u[0] = *reinterpret_cast<const uint4*>(aL + a0);
      Al.u[1] = *reinterpret_cast<const uint4*>(aL + a1);

#pragma unroll
      for (int t = 0; t < 4; ++t) {
        const size_t brow = (size_t)(ct0 + t * 16 + lr) * D_DIM + k + half * 8;
        Frag16 Bh, Bl;
        Bh.u[0] = *reinterpret_cast<const uint4*>(Xh + brow);
        Bh.u[1] = *reinterpret_cast<const uint4*>(Xh + brow + 16);
        Bl.u[0] = *reinterpret_cast<const uint4*>(Xl + brow);
        Bl.u[1] = *reinterpret_cast<const uint4*>(Xl + brow + 16);
        // dot ~= ah*bh + ah*bl + al*bh  (3xbf16 split-precision)
        acc[t] = __builtin_amdgcn_wmma_f32_16x16x32_bf16(
            false, Ah.v, false, Bh.v, (short)0, acc[t], false, false);
        acc[t] = __builtin_amdgcn_wmma_f32_16x16x32_bf16(
            false, Ah.v, false, Bl.v, (short)0, acc[t], false, false);
        acc[t] = __builtin_amdgcn_wmma_f32_16x16x32_bf16(
            false, Al.v, false, Bh.v, (short)0, acc[t], false, false);
      }
    }

    // Epilogue. C/D layout: reg v, half h -> row M = 8h + v ; col N = lr.
    float xs[8], lanepart[8];
#pragma unroll
    for (int v = 0; v < 8; ++v) {
      xs[v] = xsq[rowBase + half * 8 + v];
      lanepart[v] = 0.0f;
    }
#pragma unroll
    for (int t = 0; t < 4; ++t) {
      const int   j  = ct0 + t * 16 + lr;
      const float yj = y[j];
      const float Xs = Xsq[j];
#pragma unroll
      for (int v = 0; v < 8; ++v) {
        const int i  = rowBase + half * 8 + v;
        float d2 = fmaxf(xs[v] + Xs - 2.0f * acc[t][v], 0.0f);
        float w  = (j < i) ? __expf(-0.5f * d2) : 0.0f;  // mask j>=i
        lanepart[v] = fmaf(w, yj, lanepart[v]);
      }
    }
    // One 16-lane tree reduction per row (xor masks 1..8 stay in each half).
#pragma unroll
    for (int v = 0; v < 8; ++v) {
      float part = lanepart[v];
#pragma unroll
      for (int m = 8; m >= 1; m >>= 1) part += __shfl_xor(part, m, 32);
      if (lr == 0) atomicAdd(&srow[half * 8 + v], part);
    }
  }
  __syncthreads();
  if (tid < 16) atomicAdd(&out[rowBase + tid], srow[tid]);
}

// ===========================================================================
// Fallback path (d_ws too small for the bf16 split): exact fp32 WMMA
// (v_wmma_f32_16x16x4_f32), one 16x16 tile per wave.
// ===========================================================================
__global__ __launch_bounds__(256) void krr_wmma_f32(
    const float* __restrict__ x, const float* __restrict__ Xt,
    const float* __restrict__ y, const float* __restrict__ xsq,
    const float* __restrict__ Xsq, float* __restrict__ out) {
  const int rowBase   = blockIdx.x * 16;
  const int chunkBase = blockIdx.y * 128;
  if (chunkBase >= rowBase + 16) return;

  __shared__ float srow[16];
  const int tid = threadIdx.x;
  if (tid < 16) srow[tid] = 0.0f;
  __syncthreads();

  const int wave = tid >> 5;
  const int lane = tid & 31;
  const int half = lane >> 4;
  const int lr   = lane & 15;
  const int ct   = chunkBase + wave * 16;

  if (ct < rowBase + 16) {
    const float* aRow = x  + (size_t)(rowBase + lr) * D_DIM + half * 2;
    const float* bRow = Xt + (size_t)(ct      + lr) * D_DIM + half * 2;
    v8f acc = {};
#pragma unroll 4
    for (int kk = 0; kk < D_DIM; kk += 4) {
      v2f a = *reinterpret_cast<const v2f*>(aRow + kk);
      v2f b = *reinterpret_cast<const v2f*>(bRow + kk);
      acc = __builtin_amdgcn_wmma_f32_16x16x4_f32(
          false, a, false, b, (short)0, acc, false, false);
    }
    const int   j  = ct + lr;
    const float yj = y[j];
    const float Xs = Xsq[j];
#pragma unroll
    for (int v = 0; v < 8; ++v) {
      const int i  = rowBase + half * 8 + v;
      float d2 = fmaxf(xsq[i] + Xs - 2.0f * acc[v], 0.0f);
      float w  = (j < i) ? __expf(-0.5f * d2) : 0.0f;
      float part = w * yj;
#pragma unroll
      for (int m = 8; m >= 1; m >>= 1) part += __shfl_xor(part, m, 32);
      if (lr == 0) atomicAdd(&srow[half * 8 + v], part);
    }
  }
  __syncthreads();
  if (tid < 16) atomicAdd(&out[rowBase + tid], srow[tid]);
}

extern "C" void kernel_launch(void* const* d_in, const int* in_sizes, int n_in,
                              void* d_out, int out_size, void* d_ws,
                              size_t ws_size, hipStream_t stream) {
  const float* x  = (const float*)d_in[0];  // [M, D]
  const float* Xt = (const float*)d_in[1];  // [N, D]
  const float* y  = (const float*)d_in[2];  // [N, 1]
  float* out = (float*)d_out;               // [M, 1]

  // Workspace layout:
  //   [0)        xsq : M floats
  //   [32KB)     Xsq : N floats
  //   [64KB)     x_hi, x_lo, X_hi, X_lo : M*D / N*D bf16 each (4 MB each)
  float* xsq = (float*)d_ws;
  float* Xsq = xsq + M_DIM;
  __hip_bfloat16* xh = (__hip_bfloat16*)((char*)d_ws + 65536);
  __hip_bfloat16* xl = xh + (size_t)M_DIM * D_DIM;
  __hip_bfloat16* Xh = xl + (size_t)M_DIM * D_DIM;
  __hip_bfloat16* Xl = Xh + (size_t)N_DIM * D_DIM;
  const size_t need_split =
      65536 + 2ull * sizeof(__hip_bfloat16) * ((size_t)M_DIM + N_DIM) * D_DIM;

  zero_f32<<<(M_DIM + 255) / 256, 256, 0, stream>>>(out, M_DIM);
  row_sqnorm<<<(M_DIM + 255) / 256, 256, 0, stream>>>(x, xsq, M_DIM);
  row_sqnorm<<<(N_DIM + 255) / 256, 256, 0, stream>>>(Xt, Xsq, N_DIM);

  if (ws_size >= need_split) {
    const int nx = M_DIM * D_DIM, nX = N_DIM * D_DIM;
    split_bf16<<<(nx + 255) / 256, 256, 0, stream>>>(x, xh, xl, nx);
    split_bf16<<<(nX + 255) / 256, 256, 0, stream>>>(Xt, Xh, Xl, nX);
    dim3 grid(M_DIM / 16, N_DIM / 512);  // 512 x 16, ~half culled by mask
    krr_wmma_bf16x3<<<grid, 256, 0, stream>>>(xh, xl, Xh, Xl, y, xsq, Xsq, out);
  } else {
    dim3 grid(M_DIM / 16, N_DIM / 128);  // 512 x 64, ~half culled by mask
    krr_wmma_f32<<<grid, 256, 0, stream>>>(x, Xt, y, xsq, Xsq, out);
  }
}
